// AttnBlock_61100204753108
// MI455X (gfx1250) — matrix operand order repaired and checked by reading, not compiled, not run on hardware
//
#include <hip/hip_runtime.h>
#include <hip/hip_bf16.h>
#include <stdint.h>

typedef unsigned short u16;
typedef __bf16 v16bf __attribute__((ext_vector_type(16)));
typedef float  v8f   __attribute__((ext_vector_type(8)));

#define CC 512
#define NN 4096

__device__ __forceinline__ u16 f2bf(float x) {
  unsigned u = __float_as_uint(x);
  unsigned r = u + 0x7FFFu + ((u >> 16) & 1u);
  return (u16)(r >> 16);
}

union FragCast { uint4 u[2]; v16bf v; };

// A fragment (16x32 bf16, M = lane&15): halves K = 8h+0..7 and 16+8h+0..7
__device__ __forceinline__ v16bf loadA(const u16* base, int stride) {
  int l = threadIdx.x & 31;
  const u16* p = base + (size_t)(l & 15) * stride + 8 * (l >> 4);
  FragCast f;
  f.u[0] = *(const uint4*)(p);
  f.u[1] = *(const uint4*)(p + 16);
  return f.v;
}

// B fragment (32x16 bf16, N = lane&15): halves K = 8h+0..7 and 16+8h+0..7
__device__ __forceinline__ v16bf loadB(const u16* base, int stride) {
  int l = threadIdx.x & 31;
  const u16* p = base + (size_t)(l & 15) * stride + 8 * (l >> 4);
  FragCast f;
  f.u[0] = *(const uint4*)(p);
  f.u[1] = *(const uint4*)(p + 16);
  return f.v;
}

__device__ __forceinline__ v8f wmma_bf16(v16bf a, v16bf b, v8f c) {
  return __builtin_amdgcn_wmma_f32_16x16x32_bf16(false, a, false, b, (short)0, c,
                                                 false, false);
}

// Async DMA: 16 bytes per lane, global -> LDS, tracked by ASYNCcnt
__device__ __forceinline__ void async_cp16(const u16* g, u16* l) {
  unsigned lo = (unsigned)(uintptr_t)l;           // LDS byte offset (low 32 bits)
  unsigned long long ga = (unsigned long long)g;
  asm volatile("global_load_async_to_lds_b128 %0, %1, off"
               :: "v"(lo), "v"(ga) : "memory");
}

// ---------------- weight fp32 -> bf16 (4 x 512x512 packed consecutively) ------
__global__ void wconv_kernel(const float* a, const float* b, const float* c,
                             const float* d, u16* out) {
  int i = blockIdx.x * 256 + threadIdx.x;   // 4*262144 total
  int seg = i >> 18;
  int off = i & 262143;
  const float* src = (seg == 0) ? a : (seg == 1) ? b : (seg == 2) ? c : d;
  out[i] = f2bf(src[off]);
}

// ---------------- GroupNorm stats: one block per (b,g), 64 blocks -------------
__global__ void gn_stats_kernel(const float* x, float* stats) {
  int bg = blockIdx.x;                       // b*32+g
  const float* p = x + (size_t)bg * 16 * NN; // 16 channels * 4096 pixels
  float s = 0.f, s2 = 0.f;
  for (int i = threadIdx.x; i < 16 * NN; i += 256) {
    float v = p[i];
    s += v; s2 += v * v;
  }
  __shared__ float rs[256], rq[256];
  rs[threadIdx.x] = s; rq[threadIdx.x] = s2;
  __syncthreads();
  for (int o = 128; o > 0; o >>= 1) {
    if (threadIdx.x < o) { rs[threadIdx.x] += rs[threadIdx.x + o];
                           rq[threadIdx.x] += rq[threadIdx.x + o]; }
    __syncthreads();
  }
  if (threadIdx.x == 0) {
    float inv = 1.0f / (16.0f * NN);
    float mean = rs[0] * inv;
    float var = rq[0] * inv - mean * mean;
    stats[2 * bg] = mean;
    stats[2 * bg + 1] = rsqrtf(var + 1e-5f);
  }
}

// ---- GN apply + transpose: x[b][c][t] f32 -> hn[b][t][c] bf16 (LDS tiles) ----
__global__ void gn_apply_kernel(const float* x, const float* stats,
                                const float* gw, const float* gb, u16* hn) {
  int b = blockIdx.z;
  int c0 = blockIdx.y * 32;
  int t0 = blockIdx.x * 32;
  __shared__ float tile[32][33];
#pragma unroll
  for (int i = 0; i < 4; i++) {
    int c = c0 + threadIdx.y + 8 * i;
    int g = c >> 4;
    float mean = stats[2 * (b * 32 + g)];
    float rstd = stats[2 * (b * 32 + g) + 1];
    float v = x[((size_t)b * CC + c) * NN + t0 + threadIdx.x];
    tile[threadIdx.y + 8 * i][threadIdx.x] = (v - mean) * rstd * gw[c] + gb[c];
  }
  __syncthreads();
#pragma unroll
  for (int i = 0; i < 4; i++) {
    int t = t0 + threadIdx.y + 8 * i;
    int c = c0 + threadIdx.x;
    hn[((size_t)b * NN + t) * CC + c] = f2bf(tile[threadIdx.x][threadIdx.y + 8 * i]);
  }
}

// ---------------- Q/K GEMM: Out[t][o] = sum_c hn[t][c]*W[o][c] + bias[o] ------
__global__ void __launch_bounds__(256) qk_gemm_kernel(
    const u16* hn, const u16* wq, const u16* wk,
    const float* qb, const float* kb, u16* Q, u16* K) {
  int z = blockIdx.z;
  int b = z >> 1, which = z & 1;
  const u16* W = which ? wk : wq;
  const float* bias = which ? kb : qb;
  u16* Out = (which ? K : Q) + (size_t)b * NN * CC;
  const u16* hnB = hn + (size_t)b * NN * CC;

  int wid = threadIdx.x >> 5, l = threadIdx.x & 31, h = l >> 4;
  int tbase = blockIdx.x * 32 + (wid >> 2) * 16;
  int nbase = blockIdx.y * 256 + (wid & 3) * 64;

  v8f acc[4];
#pragma unroll
  for (int n = 0; n < 4; n++) acc[n] = (v8f){0,0,0,0,0,0,0,0};

#pragma unroll
  for (int i = 0; i < 16; i++) {
    v16bf a = loadA(hnB + (size_t)tbase * CC + i * 32, CC);
#pragma unroll
    for (int n = 0; n < 4; n++) {
      v16bf bfr = loadB(W + (size_t)(nbase + 16 * n) * CC + i * 32, CC);
      acc[n] = wmma_bf16(a, bfr, acc[n]);
    }
  }
#pragma unroll
  for (int n = 0; n < 4; n++) {
    int ocol = nbase + 16 * n + (l & 15);
    float bv = bias[ocol];
#pragma unroll
    for (int r = 0; r < 8; r++)
      Out[(size_t)(tbase + r + 8 * h) * CC + ocol] = f2bf(acc[n][r] + bv);
  }
}

// ------ V GEMM (channel-major out): Vt[o][t] = sum_c W[o][c]*hn[t][c]+vb[o] ---
__global__ void __launch_bounds__(256) v_gemm_kernel(
    const u16* hn, const u16* wv, const float* vb, u16* Vt) {
  int b = blockIdx.z;
  const u16* hnB = hn + (size_t)b * NN * CC;
  u16* VtB = Vt + (size_t)b * CC * NN;

  int wid = threadIdx.x >> 5, l = threadIdx.x & 31, h = l >> 4;
  int obase = blockIdx.x * 32 + (wid >> 2) * 16;   // M = output channel
  int tbase = blockIdx.y * 256 + (wid & 3) * 64;   // N = token

  v8f acc[4];
#pragma unroll
  for (int n = 0; n < 4; n++) acc[n] = (v8f){0,0,0,0,0,0,0,0};

#pragma unroll
  for (int i = 0; i < 16; i++) {
    v16bf a = loadA(wv + (size_t)obase * CC + i * 32, CC);
#pragma unroll
    for (int n = 0; n < 4; n++) {
      v16bf bfr = loadB(hnB + (size_t)(tbase + 16 * n) * CC + i * 32, CC);
      acc[n] = wmma_bf16(a, bfr, acc[n]);
    }
  }
#pragma unroll
  for (int r = 0; r < 8; r++) {
    int orow = obase + r + 8 * h;
    float bv = vb[orow];
#pragma unroll
    for (int n = 0; n < 4; n++) {
      int tcol = tbase + 16 * n + (l & 15);
      VtB[(size_t)orow * NN + tcol] = f2bf(acc[n][r] + bv);
    }
  }
}

// --------------------------- Flash attention ---------------------------------
// wave = 16 query rows, full d=512 f32 accumulators in VGPRs.
// K/V tiles stream through LDS via async-to-LDS DMA, double buffered:
// DMA of tile kt+1 overlaps WMMA on tile kt (ASYNCcnt pipeline).
__global__ void __launch_bounds__(256, 1) flash_kernel(
    const u16* Q, const u16* K, const u16* Vt, u16* O) {
  int b = blockIdx.y;
  size_t bo = (size_t)b * NN * CC;
  const u16* Qb = Q + bo;
  const u16* Kb = K + bo;
  const u16* Vb = Vt + bo;  // [c][t]
  u16* Ob = O + bo;         // [t][c]

  int tid = threadIdx.x;
  int wid = tid >> 5, l = tid & 31, h = l >> 4;
  int qbase = (blockIdx.x * 8 + wid) * 16;

  __shared__ __align__(16) u16 kbuf[2][32 * 512];   // 2 x 32KB : 32 keys x 512 ch
  __shared__ __align__(16) u16 vbuf[2][512 * 32];   // 2 x 32KB : 512 ch x 32 keys
  __shared__ __align__(16) u16 pbuf[8][16 * 32];    // 8KB P staging (1KB/wave)
  u16* pw = pbuf[wid];

  // issue async staging of tile kt into buffer buf (16 async ops per wave)
  auto stage_tile = [&](int kt, int buf) {
    const u16* Ksrc = Kb + (size_t)(kt * 32) * CC;
    u16* kd = kbuf[buf];
#pragma unroll
    for (int it = 0; it < 8; it++) {              // 2048 16B chunks / 256 thr
      int cidx = tid + 256 * it;
      int row = cidx >> 6, sub = cidx & 63;       // 32 rows x 64 chunks
      async_cp16(Ksrc + (size_t)row * CC + sub * 8, kd + row * 512 + sub * 8);
    }
    const u16* Vsrc = Vb + (size_t)kt * 32;
    u16* vd = vbuf[buf];
#pragma unroll
    for (int it = 0; it < 8; it++) {
      int cidx = tid + 256 * it;
      int row = cidx >> 2, sub = cidx & 3;        // 512 rows x 4 chunks
      async_cp16(Vsrc + (size_t)row * NN + sub * 8, vd + row * 32 + sub * 8);
    }
  };

  v16bf qf[16];
#pragma unroll
  for (int i = 0; i < 16; i++)
    qf[i] = loadA(Qb + (size_t)qbase * CC + i * 32, CC);

  v8f acc[32];
#pragma unroll
  for (int t = 0; t < 32; t++) acc[t] = (v8f){0,0,0,0,0,0,0,0};
  float m[8], lsum[8];
#pragma unroll
  for (int r = 0; r < 8; r++) { m[r] = -__builtin_inff(); lsum[r] = 0.f; }

  const float scale = 0.044194173824159216f;  // 512^-0.5

  stage_tile(0, 0);
  stage_tile(1, 1);

  for (int kt = 0; kt < 128; kt++) {
    int cur = kt & 1;
    // tile kt's 16 async ops were issued before tile kt+1's 16:
    // waiting ASYNCcnt<=16 guarantees tile kt is resident (<=0 on last tile).
    if (kt < 127) asm volatile("s_wait_asynccnt 16" ::: "memory");
    else          asm volatile("s_wait_asynccnt 0"  ::: "memory");
    __syncthreads();

    const u16* kc = kbuf[cur];
    v8f s0 = (v8f){0,0,0,0,0,0,0,0};
    v8f s1 = (v8f){0,0,0,0,0,0,0,0};
#pragma unroll
    for (int i = 0; i < 16; i++) {
      v16bf b0 = loadB(kc + i * 32, 512);             // keys 0..15
      v16bf b1 = loadB(kc + 16 * 512 + i * 32, 512);  // keys 16..31
      s0 = wmma_bf16(qf[i], b0, s0);
      s1 = wmma_bf16(qf[i], b1, s1);
    }
    float alpha[8];
#pragma unroll
    for (int r = 0; r < 8; r++) {
      float v0 = s0[r] * scale, v1 = s1[r] * scale;
      float mx = fmaxf(v0, v1);
#pragma unroll
      for (int off = 8; off >= 1; off >>= 1)
        mx = fmaxf(mx, __shfl_xor(mx, off, 16));
      float mn = fmaxf(m[r], mx);
      alpha[r] = __expf(m[r] - mn);
      m[r] = mn;
      float e0 = __expf(v0 - mn), e1 = __expf(v1 - mn);
      float rsum = e0 + e1;
#pragma unroll
      for (int off = 8; off >= 1; off >>= 1)
        rsum += __shfl_xor(rsum, off, 16);
      lsum[r] = lsum[r] * alpha[r] + rsum;
      // stage P (C-layout -> LDS, row = q, col = key)
      pw[(r + 8 * h) * 32 + (l & 15)] = f2bf(e0);
      pw[(r + 8 * h) * 32 + 16 + (l & 15)] = f2bf(e1);
    }
    // rescale running output accumulators
#pragma unroll
    for (int t = 0; t < 32; t++)
#pragma unroll
      for (int r = 0; r < 8; r++) acc[t][r] *= alpha[r];

    asm volatile("s_wait_dscnt 0" ::: "memory");  // LDS in-order per wave

    // reload P as A-fragment (16 q x 32 keys)
    FragCast pf;
    {
      const u16* pr = pw + (l & 15) * 32 + 8 * h;
      pf.u[0] = *(const uint4*)(pr);
      pf.u[1] = *(const uint4*)(pr + 16);
    }
    const u16* vc = vbuf[cur];
#pragma unroll
    for (int j = 0; j < 32; j++) {
      v16bf bv = loadB(vc + j * 16 * 32, 32);
      acc[j] = wmma_bf16(pf.v, bv, acc[j]);
    }
    __syncthreads();                 // all waves done with buffer `cur`
    if (kt + 2 < 128) stage_tile(kt + 2, cur);   // overlap DMA with next tile
  }

#pragma unroll
  for (int r = 0; r < 8; r++) lsum[r] = 1.0f / lsum[r];
#pragma unroll
  for (int t = 0; t < 32; t++) {
#pragma unroll
    for (int r = 0; r < 8; r++) {
      Ob[(size_t)(qbase + r + 8 * h) * CC + t * 16 + (l & 15)] =
          f2bf(acc[t][r] * lsum[r]);
    }
  }
}

// ---- proj GEMM + bias + residual: out[b][o][t] = x + pb[o] + sum_c W*Oatt ----
__global__ void __launch_bounds__(256) proj_gemm_kernel(
    const u16* Oatt, const u16* wp, const float* pb, const float* x,
    float* out) {
  int b = blockIdx.z;
  const u16* OB = Oatt + (size_t)b * NN * CC;
  size_t bofs = (size_t)b * CC * NN;

  int wid = threadIdx.x >> 5, l = threadIdx.x & 31, h = l >> 4;
  int obase = blockIdx.x * 32 + (wid >> 2) * 16;   // M = output channel
  int tbase = blockIdx.y * 256 + (wid & 3) * 64;   // N = token

  v8f acc[4];
#pragma unroll
  for (int n = 0; n < 4; n++) acc[n] = (v8f){0,0,0,0,0,0,0,0};

#pragma unroll
  for (int i = 0; i < 16; i++) {
    v16bf a = loadA(wp + (size_t)obase * CC + i * 32, CC);
#pragma unroll
    for (int n = 0; n < 4; n++) {
      v16bf bfr = loadB(OB + (size_t)(tbase + 16 * n) * CC + i * 32, CC);
      acc[n] = wmma_bf16(a, bfr, acc[n]);
    }
  }
#pragma unroll
  for (int r = 0; r < 8; r++) {
    int orow = obase + r + 8 * h;
    float bv = pb[orow];
#pragma unroll
    for (int n = 0; n < 4; n++) {
      int tcol = tbase + 16 * n + (l & 15);
      size_t idx = bofs + (size_t)orow * NN + tcol;
      out[idx] = x[idx] + bv + acc[n][r];
    }
  }
}

extern "C" void kernel_launch(void* const* d_in, const int* in_sizes, int n_in,
                              void* d_out, int out_size, void* d_ws, size_t ws_size,
                              hipStream_t stream) {
  const float* x    = (const float*)d_in[0];
  const float* gn_w = (const float*)d_in[1];
  const float* gn_b = (const float*)d_in[2];
  const float* q_w  = (const float*)d_in[3];
  const float* q_b  = (const float*)d_in[4];
  const float* k_w  = (const float*)d_in[5];
  const float* k_b  = (const float*)d_in[6];
  const float* v_w  = (const float*)d_in[7];
  const float* v_b  = (const float*)d_in[8];
  const float* p_w  = (const float*)d_in[9];
  const float* p_b  = (const float*)d_in[10];
  float* out = (float*)d_out;

  char* ws = (char*)d_ws;
  float* stats = (float*)ws;                              // 512 B
  u16* wqb = (u16*)(ws + 4096);                           // 4 x 512 KB bf16 weights
  u16* wkb = wqb + 262144;
  u16* wvb = wkb + 262144;
  u16* wpb = wvb + 262144;
  u16* hn  = (u16*)(ws + 4096 + 4 * 524288);              // [2][4096][512] bf16
  u16* Qb  = hn + 2 * NN * CC;
  u16* Kb  = Qb + 2 * NN * CC;
  u16* Vt  = Kb + 2 * NN * CC;                            // [2][512][4096] bf16
  u16* Ob  = Vt + 2 * NN * CC;                            // [2][4096][512] bf16

  wconv_kernel<<<4096, 256, 0, stream>>>(q_w, k_w, v_w, p_w, wqb);
  gn_stats_kernel<<<64, 256, 0, stream>>>(x, stats);
  gn_apply_kernel<<<dim3(128, 16, 2), dim3(32, 8), 0, stream>>>(x, stats, gn_w,
                                                                gn_b, hn);
  qk_gemm_kernel<<<dim3(128, 2, 4), 256, 0, stream>>>(hn, wqb, wkb, q_b, k_b,
                                                      Qb, Kb);
  v_gemm_kernel<<<dim3(16, 16, 2), 256, 0, stream>>>(hn, wvb, v_b, Vt);
  flash_kernel<<<dim3(32, 2), 256, 0, stream>>>(Qb, Kb, Vt, Ob);
  proj_gemm_kernel<<<dim3(16, 16, 2), 256, 0, stream>>>(Ob, wpb, p_b, x, out);
}
